// LSTM_15719580303981
// MI455X (gfx1250) — matrix operand order repaired and checked by reading, not compiled, not run on hardware
//
#include <hip/hip_runtime.h>

// -------- problem constants --------
#define HH   1024          // hidden
#define FG   4096          // 4*H (gates)
#define BB   64            // batch
#define TT   256           // timesteps
#define II   512           // layer-0 input

typedef __attribute__((ext_vector_type(16))) __bf16 v16bf;
typedef __attribute__((ext_vector_type(8)))  float  v8f;

union Frag { uint4 u[2]; v16bf v; };

__device__ __forceinline__ unsigned short f2bf(float f) {
  unsigned int u = __float_as_uint(f);
  u += 0x7fffu + ((u >> 16) & 1u);      // round-to-nearest-even
  return (unsigned short)(u >> 16);
}
__device__ __forceinline__ float sigf(float x) {
  return 1.0f / (1.0f + __expf(-x));
}
__device__ __forceinline__ v8f wmma_bf16(v16bf a, v16bf b, v8f c) {
  return __builtin_amdgcn_wmma_f32_16x16x32_bf16(false, a, false, b,
                                                 (short)0, c, false, false);
}

// -------- conversion / setup kernels --------

// Build fused [4096 x (KX+1024)] bf16 weight matrix: [Wih | Whh]
__global__ void build_wcat_kernel(const float* __restrict__ wih,
                                  const float* __restrict__ whh,
                                  int kx, unsigned short* __restrict__ wcat,
                                  size_t total) {
  size_t i = (size_t)blockIdx.x * blockDim.x + threadIdx.x;
  if (i >= total) return;
  int ld = kx + HH;
  size_t r = i / ld;
  int k = (int)(i % ld);
  float v = (k < kx) ? wih[r * (size_t)kx + k]
                     : whh[r * (size_t)HH + (k - kx)];
  wcat[i] = f2bf(v);
}

__global__ void bias_add_kernel(const float* __restrict__ a,
                                const float* __restrict__ b,
                                float* __restrict__ out, int n) {
  int i = blockIdx.x * blockDim.x + threadIdx.x;
  if (i < n) out[i] = a[i] + b[i];
}

__global__ void f2bf_kernel(const float* __restrict__ in,
                            unsigned short* __restrict__ out, size_t n) {
  size_t i = (size_t)blockIdx.x * blockDim.x + threadIdx.x;
  if (i < n) out[i] = f2bf(in[i]);
}

__global__ void init_state_kernel(const float* __restrict__ hid,
                                  const float* __restrict__ cell,
                                  float* __restrict__ h_state,
                                  float* __restrict__ c_state,
                                  unsigned short* __restrict__ hbf, int n) {
  int i = blockIdx.x * blockDim.x + threadIdx.x;
  if (i < n) {
    float h = hid[i];
    h_state[i] = h;
    c_state[i] = cell[i];
    hbf[i] = f2bf(h);
  }
}

__global__ void finalize_kernel(const float* __restrict__ h_state,
                                const float* __restrict__ c_state,
                                float* __restrict__ hn,
                                float* __restrict__ cn, int n) {
  int i = blockIdx.x * blockDim.x + threadIdx.x;
  if (i < n) { hn[i] = h_state[i]; cn[i] = c_state[i]; }
}

// -------- fused LSTM step: gates GEMM (WMMA bf16) + cell update --------
// grid = 64 blocks (one per 16 hidden columns), block = 256 threads (8 waves).
// Wave w: M-tile = w&3 (16 of 64 batch rows), gate pair = w>>2 ({i,f} or {g,o}).
// Each wave: 2 accumulator tiles, K-loop over [x_t | h_{t-1}].
template <int KX>
__global__ __launch_bounds__(256) void lstm_step_kernel(
    const unsigned short* __restrict__ xt,       // [64, KX] bf16
    const unsigned short* __restrict__ wcat,     // [4096, KX+1024] bf16
    const float* __restrict__ bias,              // [4096]
    const unsigned short* __restrict__ hbf_in,   // [64, 1024] bf16 (prev h)
    unsigned short* __restrict__ hbf_out,        // [64, 1024] bf16 (new h)
    float* __restrict__ c_state,                 // [64, 1024]
    float* __restrict__ h_state,                 // [64, 1024]
    float* __restrict__ seq_f32,                 // optional: out1[t]
    unsigned short* __restrict__ seq_bf)         // optional: out0_bf16[t]
{
  constexpr int LDW = KX + HH;
  __shared__ float G[4][BB][16];                 // 16 KB gate staging

  const int tid    = threadIdx.x;
  const int lane   = tid & 31;
  const int wave   = tid >> 5;
  const int lane15 = lane & 15;
  const int lhi    = lane >> 4;                  // half-wave select
  const int mtile  = wave & 3;
  const int gpair  = wave >> 2;
  const int hcol0  = blockIdx.x * 16;
  const int g0 = gpair * 2, g1 = g0 + 1;

  // B fragments: lane holds row (g*H + hcol0 + lane15) of Wcat,
  // 16 contiguous bf16 per K-chunk, half-wave offset +16.
  const unsigned short* w0 =
      wcat + (size_t)(g0 * HH + hcol0 + lane15) * LDW + lhi * 16;
  const unsigned short* w1 =
      wcat + (size_t)(g1 * HH + hcol0 + lane15) * LDW + lhi * 16;
  // A fragment: lane holds batch row (mtile*16 + lane15);
  // 16 contiguous bf16 per K-chunk, half-wave offset +16 (matches B).
  const int rowA = mtile * 16 + lane15;
  const unsigned short* ax = xt     + (size_t)rowA * KX + lhi * 16;
  const unsigned short* ah = hbf_in + (size_t)rowA * HH + lhi * 16;

  v8f acc0 = {}, acc1 = {};

  // ---- x_t part of K ----
  #pragma unroll 4
  for (int k = 0; k < KX; k += 32) {
    Frag A, B0, B1;
    A.u[0]  = *(const uint4*)(ax + k);
    A.u[1]  = *(const uint4*)(ax + k + 8);
    B0.u[0] = *(const uint4*)(w0 + k);
    B0.u[1] = *(const uint4*)(w0 + k + 8);
    B1.u[0] = *(const uint4*)(w1 + k);
    B1.u[1] = *(const uint4*)(w1 + k + 8);
    acc0 = wmma_bf16(A.v, B0.v, acc0);
    acc1 = wmma_bf16(A.v, B1.v, acc1);
  }
  // ---- h_{t-1} part of K ----
  #pragma unroll 4
  for (int k = 0; k < HH; k += 32) {
    Frag A, B0, B1;
    A.u[0]  = *(const uint4*)(ah + k);
    A.u[1]  = *(const uint4*)(ah + k + 8);
    B0.u[0] = *(const uint4*)(w0 + KX + k);
    B0.u[1] = *(const uint4*)(w0 + KX + k + 8);
    B1.u[0] = *(const uint4*)(w1 + KX + k);
    B1.u[1] = *(const uint4*)(w1 + KX + k + 8);
    acc0 = wmma_bf16(A.v, B0.v, acc0);
    acc1 = wmma_bf16(A.v, B1.v, acc1);
  }

  // stage gates (+bias) to LDS; C layout: VGPR r -> M = r + lhi*8, N = lane15
  const float bv0 = bias[g0 * HH + hcol0 + lane15];
  const float bv1 = bias[g1 * HH + hcol0 + lane15];
  #pragma unroll
  for (int r = 0; r < 8; ++r) {
    int m = mtile * 16 + lhi * 8 + r;
    G[g0][m][lane15] = acc0[r] + bv0;
    G[g1][m][lane15] = acc1[r] + bv1;
  }
  __syncthreads();

  // LSTM cell update for this block's 16 hidden columns, all 64 batch rows
  for (int e = tid; e < BB * 16; e += 256) {
    int b = e >> 4, n = e & 15;
    float iv = sigf (G[0][b][n]);
    float fv = sigf (G[1][b][n]);
    float gv = tanhf(G[2][b][n]);
    float ov = sigf (G[3][b][n]);
    size_t idx = (size_t)b * HH + hcol0 + n;
    float c = fv * c_state[idx] + iv * gv;
    float h = ov * tanhf(c);
    c_state[idx] = c;
    h_state[idx] = h;
    unsigned short hb = f2bf(h);
    hbf_out[idx] = hb;
    if (seq_f32) seq_f32[idx] = h;
    if (seq_bf)  seq_bf[idx]  = hb;
  }
}

// -------- host orchestration --------
extern "C" void kernel_launch(void* const* d_in, const int* in_sizes, int n_in,
                              void* d_out, int out_size, void* d_ws, size_t ws_size,
                              hipStream_t stream) {
  const float* data = (const float*)d_in[0];
  const float* hid  = (const float*)d_in[1];
  const float* cell = (const float*)d_in[2];
  const float* Wih0 = (const float*)d_in[3];
  const float* Whh0 = (const float*)d_in[4];
  const float* bih0 = (const float*)d_in[5];
  const float* bhh0 = (const float*)d_in[6];
  const float* Wih1 = (const float*)d_in[7];
  const float* Whh1 = (const float*)d_in[8];
  const float* bih1 = (const float*)d_in[9];
  const float* bhh1 = (const float*)d_in[10];
  float* out = (float*)d_out;

  char* ws = (char*)d_ws;
  size_t off = 0;
  auto alloc = [&](size_t bytes) -> void* {
    void* p = ws + off;
    off = (off + bytes + 255) & ~(size_t)255;
    return p;
  };

  unsigned short* wcat0  = (unsigned short*)alloc((size_t)FG * (II + HH) * 2); // 12 MB
  unsigned short* wcat1  = (unsigned short*)alloc((size_t)FG * (HH + HH) * 2); // 16 MB
  float*          bias0  = (float*)alloc(FG * 4);
  float*          bias1  = (float*)alloc(FG * 4);
  unsigned short* xbf    = (unsigned short*)alloc((size_t)TT * BB * II * 2);   // 16 MB
  unsigned short* out0bf = (unsigned short*)alloc((size_t)TT * BB * HH * 2);   // 32 MB
  unsigned short* hbfA   = (unsigned short*)alloc((size_t)BB * HH * 2);
  unsigned short* hbfB   = (unsigned short*)alloc((size_t)BB * HH * 2);
  float*          h_st   = (float*)alloc((size_t)BB * HH * 4);
  float*          c_st   = (float*)alloc((size_t)BB * HH * 4);

  // weight / input conversion (weights stay L2-resident as bf16 afterwards)
  {
    size_t tot = (size_t)FG * (II + HH);
    build_wcat_kernel<<<(unsigned)((tot + 255) / 256), 256, 0, stream>>>(
        Wih0, Whh0, II, wcat0, tot);
  }
  {
    size_t tot = (size_t)FG * (HH + HH);
    build_wcat_kernel<<<(unsigned)((tot + 255) / 256), 256, 0, stream>>>(
        Wih1, Whh1, HH, wcat1, tot);
  }
  bias_add_kernel<<<(FG + 255) / 256, 256, 0, stream>>>(bih0, bhh0, bias0, FG);
  bias_add_kernel<<<(FG + 255) / 256, 256, 0, stream>>>(bih1, bhh1, bias1, FG);
  {
    size_t tot = (size_t)TT * BB * II;
    f2bf_kernel<<<(unsigned)((tot + 255) / 256), 256, 0, stream>>>(data, xbf, tot);
  }

  const size_t seqN = (size_t)TT * BB * HH;
  float* hn = out + seqN;            // h_n [2,64,1024]
  float* cn = hn + 2 * BB * HH;      // c_n [2,64,1024]
  const int nState = BB * HH;

  for (int layer = 0; layer < 2; ++layer) {
    init_state_kernel<<<(nState + 255) / 256, 256, 0, stream>>>(
        hid + (size_t)layer * nState, cell + (size_t)layer * nState,
        h_st, c_st, hbfA, nState);

    const unsigned short* wcat = layer ? wcat1 : wcat0;
    const float*          bias = layer ? bias1 : bias0;

    for (int t = 0; t < TT; ++t) {
      const unsigned short* hin  = (t & 1) ? hbfB : hbfA;
      unsigned short*       hout = (t & 1) ? hbfA : hbfB;
      if (layer == 0) {
        const unsigned short* xt = xbf + (size_t)t * BB * II;
        lstm_step_kernel<II><<<64, 256, 0, stream>>>(
            xt, wcat, bias, hin, hout, c_st, h_st,
            /*seq_f32=*/nullptr, /*seq_bf=*/out0bf + (size_t)t * BB * HH);
      } else {
        const unsigned short* xt = out0bf + (size_t)t * BB * HH;
        lstm_step_kernel<HH><<<64, 256, 0, stream>>>(
            xt, wcat, bias, hin, hout, c_st, h_st,
            /*seq_f32=*/out + (size_t)t * BB * HH, /*seq_bf=*/nullptr);
      }
    }

    finalize_kernel<<<(nState + 255) / 256, 256, 0, stream>>>(
        h_st, c_st, hn + (size_t)layer * nState, cn + (size_t)layer * nState,
        nState);
  }
}
